// ApproxSVDSpectralGCN_27118423507143
// MI455X (gfx1250) — matrix operand order repaired and checked by reading, not compiled, not run on hardware
//
#include <hip/hip_runtime.h>
#include <hip/hip_bf16.h>

// ---------------------------------------------------------------------------
// ApproxSVDSpectralGCN for MI455X (gfx1250, wave32, WMMA + TDM)
//   B=64 V=1024 F=2 T=12 H=128 L=3 K=64 E=32768 OUT=12, N=B*V=65536
//   1) Laplacian build (scatter + row-norm)                 [VALU/atomics]
//   2) GRU last-state: recurrent GEMM on bf16 WMMA 16x16x32 [dominant 77 GFLOP]
//      - B fragments register-resident, fp32 state in VGPRs, LDS bf16 ping-pong
//      - x tile staged by Tensor Data Mover (tensor_load_to_lds, 6-arg form)
//      - gate nonlinearities on v_rcp_f32 / v_tanh_f32 (no IEEE div chain)
//   3) top-64 SVD via subspace iteration on f32 WMMA 16x16x4
//   4) 3 spectral layers: project/filter/back-project GEMMs on f32 WMMA
//   5) final linear (tiny, VALU)
// ---------------------------------------------------------------------------

#define B_  64
#define V_  1024
#define F_  2
#define T_  12
#define H_  128
#define K_  64
#define E_  32768
#define OUT_ 12
#define N_  (B_ * V_)

typedef __attribute__((ext_vector_type(8)))  float  v8f;
typedef __attribute__((ext_vector_type(2)))  float  v2f;
typedef __attribute__((ext_vector_type(16))) __bf16 v16bf;
typedef __attribute__((ext_vector_type(4)))  unsigned int v4u;
typedef __attribute__((ext_vector_type(8)))  int v8i;
typedef __attribute__((ext_vector_type(4)))  int v4i;

__device__ inline v8f wmma_bf16(v16bf a, v16bf b, v8f c) {
  // v_wmma_f32_16x16x32_bf16  (neg_a, A, neg_b, B, c_mod, C, reuse_a, reuse_b)
  return __builtin_amdgcn_wmma_f32_16x16x32_bf16(false, a, false, b, 0, c, false, false);
}
__device__ inline v8f wmma_f32(v2f a, v2f b, v8f c) {
  // v_wmma_f32_16x16x4_f32
  return __builtin_amdgcn_wmma_f32_16x16x4_f32(false, a, false, b, 0, c, false, false);
}

__device__ inline float fast_sigmoid(float x) {
  // v_exp_f32 + v_rcp_f32, no IEEE divide fixup chain
  return __builtin_amdgcn_rcpf(1.f + __expf(-x));
}
__device__ inline float fast_tanh(float x) {
#if defined(__has_builtin)
#if __has_builtin(__builtin_amdgcn_tanhf)
  return __builtin_amdgcn_tanhf(x);       // hardware v_tanh_f32 (CDNA5 TRANS op)
#else
  float t = __expf(2.f * x);
  return (t - 1.f) * __builtin_amdgcn_rcpf(t + 1.f);
#endif
#else
  float t = __expf(2.f * x);
  return (t - 1.f) * __builtin_amdgcn_rcpf(t + 1.f);
#endif
}

__device__ inline unsigned short bf_bits(float f) {
  unsigned u = __float_as_uint(f);
  u += 0x7FFFu + ((u >> 16) & 1u);            // round-to-nearest-even
  return (unsigned short)(u >> 16);
}

__device__ inline float block_reduce_sum(float v, float* red) {
#pragma unroll
  for (int o = 16; o > 0; o >>= 1) v += __shfl_down(v, o, 32);
  int w = threadIdx.x >> 5;
  if ((threadIdx.x & 31) == 0) red[w] = v;
  __syncthreads();
  if (threadIdx.x == 0) {
    float s = 0.f;
    for (int i = 0; i < 8; ++i) s += red[i];
    red[8] = s;
  }
  __syncthreads();
  float r = red[8];
  __syncthreads();
  return r;
}

// ---------------------------------------------------------------------------
// Laplacian construction
// ---------------------------------------------------------------------------
__global__ void zero_kernel(float* p, int n) {
  int i = blockIdx.x * 256 + threadIdx.x;
  if (i < n) p[i] = 0.f;
}

__global__ void scatter_kernel(const long long* __restrict__ ei,
                               const float* __restrict__ ew,
                               float* __restrict__ adj) {
  int e = blockIdx.x * 256 + threadIdx.x;
  if (e < E_) {
    int s = (int)ei[e];
    int d = (int)ei[E_ + e];
    atomicAdd(&adj[s * V_ + d], ew[e]);
  }
}

__global__ void diag_kernel(float* adj) {
  int i = blockIdx.x * 256 + threadIdx.x;
  if (i < V_) adj[i * V_ + i] -= 1.0f;
}

__global__ __launch_bounds__(256) void rowsum_inv_kernel(const float* __restrict__ adj,
                                                         float* __restrict__ invs) {
  __shared__ float red[16];
  int i = blockIdx.x;
  float s = 0.f;
  for (int j = threadIdx.x; j < V_; j += 256) s += adj[i * V_ + j];
  s = block_reduce_sum(s, red);
  if (threadIdx.x == 0) invs[i] = (s > 0.f) ? rsqrtf(s) : 0.f;
}

__global__ void make_lap_kernel(float* __restrict__ adj, const float* __restrict__ invs) {
  int idx = blockIdx.x * 256 + threadIdx.x;
  if (idx < V_ * V_) {
    int i = idx >> 10, j = idx & (V_ - 1);
    float a = adj[idx];
    adj[idx] = ((i == j) ? 1.f : 0.f) - invs[i] * invs[j] * a;
  }
}

// ---------------------------------------------------------------------------
// GRU: pack w_hh into per-lane bf16 WMMA B fragments (contiguous k order: k = kc*32 + 16*half + s)
// fragment id f = (g*8 + nj)*4 + kc ; per lane 16 bf16 (32B)
// ---------------------------------------------------------------------------
__global__ void pack_whh_kernel(const float* __restrict__ w_hh,
                                unsigned short* __restrict__ frag) {
  int idx = blockIdx.x * 256 + threadIdx.x;  // f*32 + lane, total 96*32
  if (idx >= 96 * 32) return;
  int f = idx >> 5, lane = idx & 31;
  int g = f >> 5, nj = (f >> 2) & 7, kc = f & 3;
  int half = lane >> 4, nl = lane & 15;
  int ncol = g * H_ + nj * 16 + nl;          // gate row of w_hh (3H x H)
#pragma unroll
  for (int j = 0; j < 8; ++j) {
    int k0 = kc * 32 + half * 16 + 2 * j;    // B[k][n] = w_hh[ncol][k]
    frag[idx * 16 + 2 * j]     = bf_bits(w_hh[ncol * H_ + k0]);
    frag[idx * 16 + 2 * j + 1] = bf_bits(w_hh[ncol * H_ + k0 + 1]);
  }
}

// GRU last hidden state. One block = 64 node-sequences, 8 waves.
// wave == nj: each wave owns output columns [wave*16, wave*16+16) of all 3 gates,
// so its 12 B fragments (3 gates x 4 k-chunks) stay register-resident for the
// whole kernel. fp32 state lives in VGPRs (each lane owns its (row,col) cells);
// the bf16 copy used as WMMA A-matrix ping-pongs in LDS (1 barrier / step).
__global__ __launch_bounds__(256) void gru_kernel(const float* __restrict__ x,
                                                  const float* __restrict__ w_ih,
                                                  const float* __restrict__ b_ih,
                                                  const float* __restrict__ b_hh,
                                                  const unsigned short* __restrict__ whh_frag,
                                                  float* __restrict__ h_out) {
  __shared__ float xs[64][F_][T_];            // 6KB, same element order as x tile
  __shared__ unsigned short hb[2][64][H_];    // 2 x 16KB bf16 state ping-pong

  int row0 = blockIdx.x * 64;
  int tid = threadIdx.x, wave = tid >> 5, lane = tid & 31;
  int half = lane >> 4, ml = lane & 15;
  int col = wave * 16 + ml;

  // ---- TDM: wave 0 DMAs the contiguous 64x24-float x tile into LDS ----
  if (wave == 0) {
    unsigned long long ga = (unsigned long long)(const void*)(x + (size_t)row0 * F_ * T_);
    unsigned lds_off = (unsigned)(unsigned long long)(void*)&xs[0][0][0];  // flat addr low 32b = LDS offset
    const unsigned nel = 64u * F_ * T_;       // 1536 elements of 4B
    v4u g0; v8i g1; v4i g2, g3; v8i g4;
    g0[0] = 1u;                                                  // count=1, user descriptor
    g0[1] = lds_off;                                             // lds_addr
    g0[2] = (unsigned)(ga & 0xFFFFFFFFull);                      // global_addr[31:0]
    g0[3] = (unsigned)((ga >> 32) & 0x01FFFFFFull) | (2u << 30); // global_addr[56:32] | type=2
    g1[0] = (int)(2u << 16);                  // data_size=4B; no multicast/pad/iterate
    g1[1] = (int)((nel & 0xFFFFu) << 16);     // tensor_dim0[15:0]
    g1[2] = (int)((nel >> 16) | (1u << 16));  // tensor_dim0[31:16] | tensor_dim1=1
    g1[3] = (int)((nel & 0xFFFFu) << 16);     // tile_dim0 = nel
    g1[4] = 0;                                // tile_dim1/2 unused
    g1[5] = (int)nel;                         // tensor_dim0_stride[31:0]
    g1[6] = 0;
    g1[7] = 0;
    g2[0] = g2[1] = g2[2] = g2[3] = 0;
    g3[0] = g3[1] = g3[2] = g3[3] = 0;
    g4[0] = g4[1] = g4[2] = g4[3] = 0;
    g4[4] = g4[5] = g4[6] = g4[7] = 0;
    // amdgpu-toolchain (clang-23) 6-arg form: (v4u, v8i, v4i, v4i, v8i, cpol)
    __builtin_amdgcn_tensor_load_to_lds(g0, g1, g2, g3, g4, 0);
    __builtin_amdgcn_s_wait_tensorcnt(0);
  }
  for (int i = tid; i < 64 * H_; i += 256) hb[0][i >> 7][i & (H_ - 1)] = 0;
  __syncthreads();

  // ---- register-resident B fragments for this wave's column slab ----
  v16bf Bfrag[3][4];
#pragma unroll
  for (int g = 0; g < 3; ++g)
#pragma unroll
    for (int kc = 0; kc < 4; ++kc) {
      int f = (g * 8 + wave) * 4 + kc;
      const uint4* bp = (const uint4*)&whh_frag[(f * 32 + lane) * 16];
      union { uint4 q4[2]; v16bf v; } bf;
      bf.q4[0] = bp[0];
      bf.q4[1] = bp[1];
      Bfrag[g][kc] = bf.v;
    }

  // per-lane gate coefficients for the fixed column
  float wr0 = w_ih[col * 2],            wr1 = w_ih[col * 2 + 1];
  float wz0 = w_ih[(H_ + col) * 2],     wz1 = w_ih[(H_ + col) * 2 + 1];
  float wn0 = w_ih[(2 * H_ + col) * 2], wn1 = w_ih[(2 * H_ + col) * 2 + 1];
  float br  = b_ih[col], bz = b_ih[H_ + col], bn = b_ih[2 * H_ + col];
  float bhr = b_hh[col], bhz = b_hh[H_ + col], bhn = b_hh[2 * H_ + col];

  float hreg[4][8];
#pragma unroll
  for (int mi = 0; mi < 4; ++mi)
#pragma unroll
    for (int j = 0; j < 8; ++j) hreg[mi][j] = 0.f;

  int cur = 0;
  for (int t = 0; t < T_; ++t) {
#pragma unroll
    for (int mi = 0; mi < 4; ++mi) {
      v8f aR, aZ, aN;
#pragma unroll
      for (int j = 0; j < 8; ++j) { aR[j] = 0.f; aZ[j] = 0.f; aN[j] = 0.f; }
#pragma unroll
      for (int kc = 0; kc < 4; ++kc) {
        union { unsigned int u[8]; v16bf v; } af;
#pragma unroll
        for (int j = 0; j < 8; ++j) {
          int kb = kc * 32 + half * 16 + (j << 1);
          af.u[j] = *(const unsigned int*)&hb[cur][mi * 16 + ml][kb];
        }
        aR = wmma_bf16(af.v, Bfrag[0][kc], aR);
        aZ = wmma_bf16(af.v, Bfrag[1][kc], aZ);
        aN = wmma_bf16(af.v, Bfrag[2][kc], aN);
      }
#pragma unroll
      for (int j = 0; j < 8; ++j) {
        int row = mi * 16 + half * 8 + j;
        float x0 = xs[row][0][t], x1 = xs[row][1][t];
        float gr = x0 * wr0 + x1 * wr1 + br + aR[j] + bhr;
        float gz = x0 * wz0 + x1 * wz1 + bz + aZ[j] + bhz;
        float xn = x0 * wn0 + x1 * wn1 + bn;
        float r = fast_sigmoid(gr);
        float z = fast_sigmoid(gz);
        float nc = fast_tanh(xn + r * (aN[j] + bhn));
        float hnew = (1.f - z) * nc + z * hreg[mi][j];
        hreg[mi][j] = hnew;
        hb[cur ^ 1][row][col] = bf_bits(hnew);
      }
    }
    __syncthreads();
    cur ^= 1;
  }

#pragma unroll
  for (int mi = 0; mi < 4; ++mi)
#pragma unroll
    for (int j = 0; j < 8; ++j)
      h_out[(row0 + mi * 16 + half * 8 + j) * H_ + col] = hreg[mi][j];
}

// ---------------------------------------------------------------------------
// Subspace iteration GEMMs (f32 WMMA 16x16x4): C(1024x64) = op(A,1024x1024) @ B(1024x64)
// ---------------------------------------------------------------------------
__global__ __launch_bounds__(256) void gemm_nn_64(const float* __restrict__ A,
                                                  const float* __restrict__ Bm,
                                                  float* __restrict__ C) {
  __shared__ float As[128][33];
  __shared__ float Bs[32][K_];
  int row0 = blockIdx.x * 128;
  int tid = threadIdx.x, wave = tid >> 5, lane = tid & 31;
  int half = lane >> 4, ml = lane & 15;
  v8f acc[4];
#pragma unroll
  for (int nj = 0; nj < 4; ++nj)
#pragma unroll
    for (int j = 0; j < 8; ++j) acc[nj][j] = 0.f;

  for (int kc = 0; kc < V_ / 32; ++kc) {
    __syncthreads();
#pragma unroll
    for (int qd = 0; qd < 4; ++qd) {
      int tI = tid + qd * 256;
      int m = tI >> 3, kk4 = (tI & 7) * 4;
      float4 v = *(const float4*)&A[(row0 + m) * V_ + kc * 32 + kk4];
      As[m][kk4] = v.x; As[m][kk4 + 1] = v.y; As[m][kk4 + 2] = v.z; As[m][kk4 + 3] = v.w;
    }
#pragma unroll
    for (int qd = 0; qd < 2; ++qd) {
      int tI = tid + qd * 256;
      int kk = tI >> 4, n4 = (tI & 15) * 4;
      float4 v = *(const float4*)&Bm[(kc * 32 + kk) * K_ + n4];
      Bs[kk][n4] = v.x; Bs[kk][n4 + 1] = v.y; Bs[kk][n4 + 2] = v.z; Bs[kk][n4 + 3] = v.w;
    }
    if (kc + 1 < V_ / 32)
      __builtin_prefetch(&A[(row0 + (tid >> 3)) * V_ + (kc + 1) * 32], 0, 1);
    __syncthreads();
#pragma unroll
    for (int ks = 0; ks < 8; ++ks) {
      v2f a;
      a[0] = As[wave * 16 + ml][ks * 4 + half * 2];
      a[1] = As[wave * 16 + ml][ks * 4 + half * 2 + 1];
#pragma unroll
      for (int nj = 0; nj < 4; ++nj) {
        v2f b;
        b[0] = Bs[ks * 4 + half * 2][nj * 16 + ml];
        b[1] = Bs[ks * 4 + half * 2 + 1][nj * 16 + ml];
        acc[nj] = wmma_f32(a, b, acc[nj]);
      }
    }
  }
#pragma unroll
  for (int nj = 0; nj < 4; ++nj)
#pragma unroll
    for (int j = 0; j < 8; ++j)
      C[(row0 + wave * 16 + half * 8 + j) * K_ + nj * 16 + ml] = acc[nj][j];
}

__global__ __launch_bounds__(256) void gemm_tn_64(const float* __restrict__ A,
                                                  const float* __restrict__ Bm,
                                                  float* __restrict__ C) {
  __shared__ float As[128][33];
  __shared__ float Bs[32][K_];
  int row0 = blockIdx.x * 128;
  int tid = threadIdx.x, wave = tid >> 5, lane = tid & 31;
  int half = lane >> 4, ml = lane & 15;
  v8f acc[4];
#pragma unroll
  for (int nj = 0; nj < 4; ++nj)
#pragma unroll
    for (int j = 0; j < 8; ++j) acc[nj][j] = 0.f;

  for (int kc = 0; kc < V_ / 32; ++kc) {
    __syncthreads();
#pragma unroll
    for (int qd = 0; qd < 4; ++qd) {
      int tI = tid + qd * 256;
      int kk = tI >> 5, m4 = (tI & 31) * 4;
      float4 v = *(const float4*)&A[(kc * 32 + kk) * V_ + row0 + m4];  // A^T tile
      As[m4][kk] = v.x; As[m4 + 1][kk] = v.y; As[m4 + 2][kk] = v.z; As[m4 + 3][kk] = v.w;
    }
#pragma unroll
    for (int qd = 0; qd < 2; ++qd) {
      int tI = tid + qd * 256;
      int kk = tI >> 4, n4 = (tI & 15) * 4;
      float4 v = *(const float4*)&Bm[(kc * 32 + kk) * K_ + n4];
      Bs[kk][n4] = v.x; Bs[kk][n4 + 1] = v.y; Bs[kk][n4 + 2] = v.z; Bs[kk][n4 + 3] = v.w;
    }
    if (kc + 1 < V_ / 32)
      __builtin_prefetch(&A[((kc + 1) * 32 + (tid >> 5)) * V_ + row0], 0, 1);
    __syncthreads();
#pragma unroll
    for (int ks = 0; ks < 8; ++ks) {
      v2f a;
      a[0] = As[wave * 16 + ml][ks * 4 + half * 2];
      a[1] = As[wave * 16 + ml][ks * 4 + half * 2 + 1];
#pragma unroll
      for (int nj = 0; nj < 4; ++nj) {
        v2f b;
        b[0] = Bs[ks * 4 + half * 2][nj * 16 + ml];
        b[1] = Bs[ks * 4 + half * 2 + 1][nj * 16 + ml];
        acc[nj] = wmma_f32(a, b, acc[nj]);
      }
    }
  }
#pragma unroll
  for (int nj = 0; nj < 4; ++nj)
#pragma unroll
    for (int j = 0; j < 8; ++j)
      C[(row0 + wave * 16 + half * 8 + j) * K_ + nj * 16 + ml] = acc[nj][j];
}

__global__ void init_q_kernel(float* Q) {
  int i = blockIdx.x * 256 + threadIdx.x;
  if (i < V_ * K_) {
    unsigned u = (unsigned)i * 2654435761u;
    u ^= u >> 13; u *= 0x85ebca6bu; u ^= u >> 16;
    Q[i] = (float)(u & 0xFFFFFFu) * (1.0f / 16777216.0f) - 0.5f;
  }
}

// Modified Gram-Schmidt on the 64 columns of Q (1024x64), single block.
__global__ __launch_bounds__(256) void ortho_kernel(float* __restrict__ Q) {
  __shared__ float red[16];
  int tid = threadIdx.x;
  for (int k = 0; k < K_; ++k) {
    float s = 0.f;
    for (int v = tid; v < V_; v += 256) { float q = Q[v * K_ + k]; s += q * q; }
    s = block_reduce_sum(s, red);
    float inv = rsqrtf(fmaxf(s, 1e-20f));
    for (int v = tid; v < V_; v += 256) Q[v * K_ + k] *= inv;
    __syncthreads();
    for (int j = k + 1; j < K_; ++j) {
      float d = 0.f;
      for (int v = tid; v < V_; v += 256) d += Q[v * K_ + j] * Q[v * K_ + k];
      d = block_reduce_sum(d, red);
      for (int v = tid; v < V_; v += 256) Q[v * K_ + j] -= d * Q[v * K_ + k];
      __syncthreads();
    }
  }
}

// svals[k] = ||Y[:,k]||, Ul[:,k] = Y[:,k]/svals[k]
__global__ __launch_bounds__(256) void svals_kernel(const float* __restrict__ Y,
                                                    float* __restrict__ svals,
                                                    float* __restrict__ Ul) {
  __shared__ float red[16];
  int k = blockIdx.x;
  float s = 0.f;
  for (int v = threadIdx.x; v < V_; v += 256) { float y = Y[v * K_ + k]; s += y * y; }
  s = block_reduce_sum(s, red);
  float nrm = sqrtf(s);
  if (threadIdx.x == 0) svals[k] = nrm;
  float inv = (nrm > 1e-12f) ? 1.f / nrm : 0.f;
  for (int v = threadIdx.x; v < V_; v += 256) Ul[v * K_ + k] = Y[v * K_ + k] * inv;
}

// ---------------------------------------------------------------------------
// Spectral conv GEMMs (f32 WMMA)
// ---------------------------------------------------------------------------
// spec[b](64x128) = svecs^T(64x1024) @ h[b](1024x128)
__global__ __launch_bounds__(256) void spec_kernel(const float* __restrict__ svecs,
                                                   const float* __restrict__ h,
                                                   float* __restrict__ spec) {
  __shared__ float As[64][65];
  __shared__ float Bs[64][H_];
  int b = blockIdx.x;
  int tid = threadIdx.x, wave = tid >> 5, lane = tid & 31;
  int half = lane >> 4, ml = lane & 15;
  v8f acc[4];
#pragma unroll
  for (int p = 0; p < 4; ++p)
#pragma unroll
    for (int j = 0; j < 8; ++j) acc[p][j] = 0.f;

  for (int kc = 0; kc < V_ / 64; ++kc) {
    __syncthreads();
    for (int qd = 0; qd < 16; ++qd) {
      int tI = tid + qd * 256;
      int m = tI & 63, kk = tI >> 6;
      As[m][kk] = svecs[(kc * 64 + kk) * K_ + m];  // svecs^T
    }
#pragma unroll
    for (int qd = 0; qd < 8; ++qd) {
      int tI = tid + qd * 256;
      int kk = tI >> 5, n4 = (tI & 31) * 4;
      float4 v = *(const float4*)&h[(b * V_ + kc * 64 + kk) * H_ + n4];
      Bs[kk][n4] = v.x; Bs[kk][n4 + 1] = v.y; Bs[kk][n4 + 2] = v.z; Bs[kk][n4 + 3] = v.w;
    }
    __syncthreads();
#pragma unroll
    for (int p = 0; p < 4; ++p) {
      int q = wave * 4 + p;
      int mi = q >> 3, nj = q & 7;
#pragma unroll
      for (int ks = 0; ks < 16; ++ks) {
        v2f a, bb;
        a[0] = As[mi * 16 + ml][ks * 4 + half * 2];
        a[1] = As[mi * 16 + ml][ks * 4 + half * 2 + 1];
        bb[0] = Bs[ks * 4 + half * 2][nj * 16 + ml];
        bb[1] = Bs[ks * 4 + half * 2 + 1][nj * 16 + ml];
        acc[p] = wmma_f32(a, bb, acc[p]);
      }
    }
  }
#pragma unroll
  for (int p = 0; p < 4; ++p) {
    int q = wave * 4 + p;
    int mi = q >> 3, nj = q & 7;
#pragma unroll
    for (int j = 0; j < 8; ++j)
      spec[(b * K_ + mi * 16 + half * 8 + j) * H_ + nj * 16 + ml] = acc[p][j];
  }
}

// filt[b](64x128) = (spec[b](64x128) @ w(128x128)) * svals[row]
__global__ __launch_bounds__(256) void filt_kernel(const float* __restrict__ spec,
                                                   const float* __restrict__ w,
                                                   const float* __restrict__ svals,
                                                   float* __restrict__ filt) {
  __shared__ float As[64][H_];
  __shared__ float Bs[32][H_];
  int b = blockIdx.x;
  int tid = threadIdx.x, wave = tid >> 5, lane = tid & 31;
  int half = lane >> 4, ml = lane & 15;
  v8f acc[4];
#pragma unroll
  for (int p = 0; p < 4; ++p)
#pragma unroll
    for (int j = 0; j < 8; ++j) acc[p][j] = 0.f;

#pragma unroll
  for (int qd = 0; qd < 8; ++qd) {  // spec[b] once: 64x128
    int tI = tid + qd * 256;
    int m = tI >> 5, n4 = (tI & 31) * 4;
    float4 v = *(const float4*)&spec[(b * K_ + m) * H_ + n4];
    As[m][n4] = v.x; As[m][n4 + 1] = v.y; As[m][n4 + 2] = v.z; As[m][n4 + 3] = v.w;
  }
  for (int kc = 0; kc < 4; ++kc) {
    __syncthreads();
#pragma unroll
    for (int qd = 0; qd < 4; ++qd) {
      int tI = tid + qd * 256;
      int kk = tI >> 5, n4 = (tI & 31) * 4;
      float4 v = *(const float4*)&w[(kc * 32 + kk) * H_ + n4];
      Bs[kk][n4] = v.x; Bs[kk][n4 + 1] = v.y; Bs[kk][n4 + 2] = v.z; Bs[kk][n4 + 3] = v.w;
    }
    __syncthreads();
#pragma unroll
    for (int p = 0; p < 4; ++p) {
      int q = wave * 4 + p;
      int mi = q >> 3, nj = q & 7;
#pragma unroll
      for (int ks = 0; ks < 8; ++ks) {
        v2f a, bb;
        a[0] = As[mi * 16 + ml][kc * 32 + ks * 4 + half * 2];
        a[1] = As[mi * 16 + ml][kc * 32 + ks * 4 + half * 2 + 1];
        bb[0] = Bs[ks * 4 + half * 2][nj * 16 + ml];
        bb[1] = Bs[ks * 4 + half * 2 + 1][nj * 16 + ml];
        acc[p] = wmma_f32(a, bb, acc[p]);
      }
    }
  }
#pragma unroll
  for (int p = 0; p < 4; ++p) {
    int q = wave * 4 + p;
    int mi = q >> 3, nj = q & 7;
#pragma unroll
    for (int j = 0; j < 8; ++j) {
      int row = mi * 16 + half * 8 + j;
      filt[(b * K_ + row) * H_ + nj * 16 + ml] = acc[p][j] * svals[row];
    }
  }
}

// conv[b](1024x128) (+)= svecs(1024x64) @ filt[b](64x128)
__global__ __launch_bounds__(256) void proj_kernel(const float* __restrict__ svecs,
                                                   const float* __restrict__ filt,
                                                   float* __restrict__ conv,
                                                   int addto) {
  __shared__ float As[128][33];
  __shared__ float Bs[32][H_];
  int mt = blockIdx.x, b = blockIdx.y;
  int row0 = mt * 128;
  int tid = threadIdx.x, wave = tid >> 5, lane = tid & 31;
  int half = lane >> 4, ml = lane & 15;
  v8f acc[8];
#pragma unroll
  for (int nj = 0; nj < 8; ++nj)
#pragma unroll
    for (int j = 0; j < 8; ++j)
      acc[nj][j] = addto ? conv[(b * V_ + row0 + wave * 16 + half * 8 + j) * H_ + nj * 16 + ml] : 0.f;

  for (int kc = 0; kc < 2; ++kc) {
    __syncthreads();
#pragma unroll
    for (int qd = 0; qd < 4; ++qd) {
      int tI = tid + qd * 256;
      int m = tI >> 3, k4 = (tI & 7) * 4;
      float4 v = *(const float4*)&svecs[(row0 + m) * K_ + kc * 32 + k4];
      As[m][k4] = v.x; As[m][k4 + 1] = v.y; As[m][k4 + 2] = v.z; As[m][k4 + 3] = v.w;
    }
#pragma unroll
    for (int qd = 0; qd < 4; ++qd) {
      int tI = tid + qd * 256;
      int kk = tI >> 5, n4 = (tI & 31) * 4;
      float4 v = *(const float4*)&filt[(b * K_ + kc * 32 + kk) * H_ + n4];
      Bs[kk][n4] = v.x; Bs[kk][n4 + 1] = v.y; Bs[kk][n4 + 2] = v.z; Bs[kk][n4 + 3] = v.w;
    }
    __syncthreads();
#pragma unroll
    for (int ks = 0; ks < 8; ++ks) {
      v2f a;
      a[0] = As[wave * 16 + ml][ks * 4 + half * 2];
      a[1] = As[wave * 16 + ml][ks * 4 + half * 2 + 1];
#pragma unroll
      for (int nj = 0; nj < 8; ++nj) {
        v2f bb;
        bb[0] = Bs[ks * 4 + half * 2][nj * 16 + ml];
        bb[1] = Bs[ks * 4 + half * 2 + 1][nj * 16 + ml];
        acc[nj] = wmma_f32(a, bb, acc[nj]);
      }
    }
  }
#pragma unroll
  for (int nj = 0; nj < 8; ++nj)
#pragma unroll
    for (int j = 0; j < 8; ++j)
      conv[(b * V_ + row0 + wave * 16 + half * 8 + j) * H_ + nj * 16 + ml] = acc[nj][j];
}

__global__ void relu_add_kernel(const float* __restrict__ conv, float* __restrict__ h) {
  int i = blockIdx.x * 256 + threadIdx.x;
  if (i < N_ * H_) h[i] += fmaxf(conv[i], 0.f);
}

__global__ void linear_kernel(const float* __restrict__ h,
                              const float* __restrict__ lin_w,
                              const float* __restrict__ lin_b,
                              float* __restrict__ out) {
  int i = blockIdx.x * 256 + threadIdx.x;
  if (i < N_ * OUT_) {
    int row = i / OUT_, o = i % OUT_;
    float s = lin_b[o];
    for (int c = 0; c < H_; ++c) s += h[row * H_ + c] * lin_w[o * H_ + c];
    out[i] = s;
  }
}

// ---------------------------------------------------------------------------
extern "C" void kernel_launch(void* const* d_in, const int* in_sizes, int n_in,
                              void* d_out, int out_size, void* d_ws, size_t ws_size,
                              hipStream_t stream) {
  const float*     x      = (const float*)d_in[0];
  const long long* ei     = (const long long*)d_in[1];
  const float*     ew     = (const float*)d_in[2];
  const float*     w_ih   = (const float*)d_in[3];
  const float*     w_hh   = (const float*)d_in[4];
  const float*     b_ih   = (const float*)d_in[5];
  const float*     b_hh   = (const float*)d_in[6];
  const float*     conv_w = (const float*)d_in[7];
  const float*     lin_w  = (const float*)d_in[8];
  const float*     lin_b  = (const float*)d_in[9];
  float* out = (float*)d_out;
  float* ws  = (float*)d_ws;

  size_t o = 0;
  float* hbuf = ws + o; o += (size_t)N_ * H_;       // 32MB
  float* lap  = ws + o; o += (size_t)V_ * V_;       // 4MB
  float* invs = ws + o; o += V_;
  float* Qb   = ws + o; o += (size_t)V_ * K_;
  float* Yb   = ws + o; o += (size_t)V_ * K_;
  float* Ul   = ws + o; o += (size_t)V_ * K_;
  float* sval = ws + o; o += K_;
  unsigned short* frag = (unsigned short*)(ws + o); o += 96 * 32 * 16 / 2;
  float* specb = ws + o; o += (size_t)B_ * K_ * H_;
  float* filtb = ws + o; o += (size_t)B_ * K_ * H_;
  float* convb = ws + o; o += (size_t)N_ * H_;      // 32MB

  // 1) Laplacian
  zero_kernel<<<(V_ * V_ + 255) / 256, 256, 0, stream>>>(lap, V_ * V_);
  scatter_kernel<<<(E_ + 255) / 256, 256, 0, stream>>>(ei, ew, lap);
  diag_kernel<<<(V_ + 255) / 256, 256, 0, stream>>>(lap);
  rowsum_inv_kernel<<<V_, 256, 0, stream>>>(lap, invs);
  make_lap_kernel<<<(V_ * V_ + 255) / 256, 256, 0, stream>>>(lap, invs);

  // 2) GRU (dominant compute, bf16 WMMA + TDM-staged inputs)
  pack_whh_kernel<<<12, 256, 0, stream>>>(w_hh, frag);
  gru_kernel<<<N_ / 64, 256, 0, stream>>>(x, w_ih, b_ih, b_hh, frag, hbuf);

  // 3) Approx top-64 SVD by subspace iteration (f32 WMMA)
  init_q_kernel<<<(V_ * K_ + 255) / 256, 256, 0, stream>>>(Qb);
  for (int it = 0; it < 5; ++it) {
    gemm_nn_64<<<V_ / 128, 256, 0, stream>>>(lap, Qb, Yb);   // Y = L Q
    gemm_tn_64<<<V_ / 128, 256, 0, stream>>>(lap, Yb, Qb);   // Q = L^T Y
    ortho_kernel<<<1, 256, 0, stream>>>(Qb);
  }
  gemm_nn_64<<<V_ / 128, 256, 0, stream>>>(lap, Qb, Yb);     // Y = L Vr
  svals_kernel<<<K_, 256, 0, stream>>>(Yb, sval, Ul);        // S, Ul

  // 4) Spectral GCN layers
  for (int l = 0; l < 3; ++l) {
    const float* w = conv_w + (size_t)l * H_ * H_;
    spec_kernel<<<B_, 256, 0, stream>>>(Ul, hbuf, specb);
    filt_kernel<<<B_, 256, 0, stream>>>(specb, w, sval, filtb);
    proj_kernel<<<dim3(V_ / 128, B_), 256, 0, stream>>>(Ul, filtb, convb, 0);
    spec_kernel<<<B_, 256, 0, stream>>>(Qb, hbuf, specb);
    filt_kernel<<<B_, 256, 0, stream>>>(specb, w, sval, filtb);
    proj_kernel<<<dim3(V_ / 128, B_), 256, 0, stream>>>(Qb, filtb, convb, 1);
    relu_add_kernel<<<(N_ * H_ + 255) / 256, 256, 0, stream>>>(convb, hbuf);
  }

  // 5) Output projection
  linear_kernel<<<(N_ * OUT_ + 255) / 256, 256, 0, stream>>>(hbuf, lin_w, lin_b, out);
}
